// MultiHeadAttention_1400159338542
// MI455X (gfx1250) — matrix operand order repaired and checked by reading, not compiled, not run on hardware
//
#include <hip/hip_runtime.h>
#include <math.h>

// ---------------------------------------------------------------------------
// Single-head attention for B=4, L=2048, C=512 (fp32 in/out) on gfx1250.
//
// Pipeline (all GEMMs on v_wmma_f32_16x16x32_f16, fp32 accumulate):
//   1. Qf16 = (query @ Wq + bq)           -> f16 [B,L,C]
//   2. Ktf16 = (key @ Wk + bk)^T          -> f16 [B,C,L]  (pre-transposed)
//   3. Vf16 = (value @ Wv + bv)           -> f16 [B,L,C]
//   4. S    = scale * Qf16 @ Ktf16        -> f32 [B,L,L]  (64 MB, L2-resident)
//   5. P    = softmax_rows(S)             -> f16 [B,L,L]  (32 MB)
//   6. Of16 = P @ Vf16                    -> f16 [B,L,C]
//   7. out  = Of16 @ Wo + bo              -> f32 [B,L,C]
//
// GEMM: block tile 128x128x32, 8 waves (4x2), wave tile 32x64 (2x4 WMMA
// accumulators), ping-pong double-buffered LDS (1 barrier / K-step).
// Workspace: 4*8MB (f16 tensors) + 64MB (S) + 32MB (P) = 128 MB.
// ---------------------------------------------------------------------------

typedef __attribute__((ext_vector_type(16))) _Float16 v16h;
typedef __attribute__((ext_vector_type(8)))  _Float16 v8h;
typedef __attribute__((ext_vector_type(4)))  _Float16 v4h;
typedef __attribute__((ext_vector_type(8)))  float    v8f;
typedef __attribute__((ext_vector_type(4)))  float    v4f;

#define BM 128
#define BN 128
#define BK 32
#define APAD 8
#define BPAD 8

// Tiled GEMM: Out[M,N] = scale * A[M,K] @ Bm[K,N] (+ bias[N])
// AF16/BF16: input dtype (0 = fp32 converted to f16 during LDS staging).
// OUTMODE: 0 = f32 row-major, 1 = f16 row-major, 2 = f16 per-batch transposed.
template<int AF16, int BF16, int OUTMODE, int BIAS>
__global__ __launch_bounds__(256)
void gemm_wmma(const void* __restrict__ Aq, const void* __restrict__ Bq,
               const float* __restrict__ bias, void* __restrict__ Outq,
               int lda, int ldb, int ldo, int Kdim, float scale,
               long sA, long sB, long sO, int transL)
{
  __shared__ _Float16 As[2][BM][BK + APAD];  // 2 x 128 x 40 f16
  __shared__ _Float16 Bs[2][BK][BN + BPAD];  // 2 x  32 x 136 f16

  const int t    = threadIdx.x;
  const int lane = t & 31;
  const int wid  = t >> 5;
  const int wm   = (wid & 3) * 32;   // 4 waves along M
  const int wn   = (wid >> 2) * 64;  // 2 waves along N, 64 cols each
  const int m0   = blockIdx.x * BM;
  const int n0   = blockIdx.y * BN;
  const int bz   = blockIdx.z;

  // staging assignment: A: thread -> (row ar, 16-col half); B: thread -> (row br, 16-col chunk)
  const int ar = t >> 1;         // 0..127
  const int ac = (t & 1) * 16;   // 0 or 16
  const int br = t >> 3;         // 0..31
  const int bc = (t & 7) * 16;   // 0..112

  v8f acc[2][4] = {};
  v8h sa0, sa1, sb0, sb1;        // staged data (registers) for next buffer

  // global load (+f16 convert) of the k0 tile into staging registers
  auto gload = [&](int k0) {
    if (AF16) {
      const _Float16* a = (const _Float16*)Aq + (long)bz * sA + (long)(m0 + ar) * lda + k0 + ac;
      sa0 = *(const v8h*)(a);
      sa1 = *(const v8h*)(a + 8);
    } else {
      const float* a = (const float*)Aq + (long)bz * sA + (long)(m0 + ar) * lda + k0 + ac;
      v4f x0 = *(const v4f*)(a);
      v4f x1 = *(const v4f*)(a + 4);
      v4f x2 = *(const v4f*)(a + 8);
      v4f x3 = *(const v4f*)(a + 12);
      #pragma unroll
      for (int j = 0; j < 4; ++j) {
        sa0[j]     = (_Float16)x0[j];
        sa0[4 + j] = (_Float16)x1[j];
        sa1[j]     = (_Float16)x2[j];
        sa1[4 + j] = (_Float16)x3[j];
      }
    }
    if (BF16) {
      const _Float16* b = (const _Float16*)Bq + (long)bz * sB + (long)(k0 + br) * ldb + n0 + bc;
      sb0 = *(const v8h*)(b);
      sb1 = *(const v8h*)(b + 8);
    } else {
      const float* b = (const float*)Bq + (long)bz * sB + (long)(k0 + br) * ldb + n0 + bc;
      v4f x0 = *(const v4f*)(b);
      v4f x1 = *(const v4f*)(b + 4);
      v4f x2 = *(const v4f*)(b + 8);
      v4f x3 = *(const v4f*)(b + 12);
      #pragma unroll
      for (int j = 0; j < 4; ++j) {
        sb0[j]     = (_Float16)x0[j];
        sb0[4 + j] = (_Float16)x1[j];
        sb1[j]     = (_Float16)x2[j];
        sb1[4 + j] = (_Float16)x3[j];
      }
    }
  };
  // write staged registers into LDS buffer `buf`
  auto sstore = [&](int buf) {
    *(v8h*)&As[buf][ar][ac]      = sa0;
    *(v8h*)&As[buf][ar][ac + 8]  = sa1;
    *(v8h*)&Bs[buf][br][bc]      = sb0;
    *(v8h*)&Bs[buf][br][bc + 8]  = sb1;
  };

  // prologue: fill buffer 0
  gload(0);
  sstore(0);
  __syncthreads();

  int buf = 0;
  for (int k0 = 0; k0 < Kdim; k0 += BK) {
    const bool more = (k0 + BK) < Kdim;
    // issue next tile's global loads early (hidden under the WMMA chain)
    if (more) gload(k0 + BK);
    if (k0 + 2 * BK < Kdim) {  // L2 prefetch two tiles ahead (global_prefetch_b8)
      const char* pa = (const char*)Aq +
          ((long)bz * sA + (long)(m0 + ar) * lda + (k0 + 2 * BK) + ac) * (AF16 ? 2 : 4);
      const char* pb = (const char*)Bq +
          ((long)bz * sB + (long)(k0 + 2 * BK + br) * ldb + n0 + bc) * (BF16 ? 2 : 4);
      __builtin_prefetch(pa, 0, 1);
      __builtin_prefetch(pb, 0, 1);
    }

    // ---- fragments per ISA-documented wave32 WMMA layouts ----
    // A (16x32 f16): lanes 0-15 -> K 0..7 / 16..23, lanes 16-31 -> K 8..15 / 24..31
    const int arow = lane & 15;
    const int kh   = (lane >> 4) * 8;
    v16h afrag[2], bfrag[4];
    #pragma unroll
    for (int i = 0; i < 2; ++i) {
      v8h lo = *(const v8h*)&As[buf][wm + i * 16 + arow][kh];
      v8h hi = *(const v8h*)&As[buf][wm + i * 16 + arow][kh + 16];
      afrag[i] = __builtin_shufflevector(lo, hi, 0, 1, 2, 3, 4, 5, 6, 7,
                                                 8, 9, 10, 11, 12, 13, 14, 15);
    }
    // B (32x16 f16): lane%16 = N col; lanes 0-15 -> K 0..7 / 16..23, lanes 16-31 -> K 8..15 / 24..31
    #pragma unroll
    for (int j = 0; j < 4; ++j) {
      #pragma unroll
      for (int s = 0; s < 8; ++s) {
        bfrag[j][s]     = Bs[buf][kh + s][wn + j * 16 + arow];
        bfrag[j][8 + s] = Bs[buf][kh + 16 + s][wn + j * 16 + arow];
      }
    }
    #pragma unroll
    for (int i = 0; i < 2; ++i)
      #pragma unroll
      for (int j = 0; j < 4; ++j)
        acc[i][j] = __builtin_amdgcn_wmma_f32_16x16x32_f16(
            false, afrag[i], false, bfrag[j], (short)0, acc[i][j], false, false);

    // store next tile into the other buffer, then one barrier per K-step
    if (more) sstore(buf ^ 1);
    __syncthreads();
    buf ^= 1;
  }

  // ---- epilogue: C/D layout -> VGPR r: lanes 0-15 M=r, lanes 16-31 M=8+r ----
  const int rb = (lane >> 4) * 8;
  const int nl = lane & 15;
  #pragma unroll
  for (int i = 0; i < 2; ++i) {
    #pragma unroll
    for (int j = 0; j < 4; ++j) {
      const int gc = n0 + wn + j * 16 + nl;
      const float bv = BIAS ? bias[gc] : 0.0f;
      #pragma unroll
      for (int r = 0; r < 8; ++r) {
        const float v  = acc[i][j][r] * scale + bv;
        const int   gr = m0 + wm + i * 16 + rb + r;
        if (OUTMODE == 0) {
          ((float*)Outq)[(long)bz * sO + (long)gr * ldo + gc] = v;
        } else if (OUTMODE == 1) {
          ((_Float16*)Outq)[(long)bz * sO + (long)gr * ldo + gc] = (_Float16)v;
        } else {  // per-batch transposed: out[b][gc][l]
          const int b = gr / transL;
          const int l = gr % transL;
          ((_Float16*)Outq)[(long)b * sO + (long)gc * transL + l] = (_Float16)v;
        }
      }
    }
  }
}

// Row softmax: one block per row of length n=2048 (8 fp32 per thread, in-register).
__global__ __launch_bounds__(256)
void softmax_rows(const float* __restrict__ S, _Float16* __restrict__ P, int n)
{
  const long row = blockIdx.x;
  const float* s = S + row * (long)n;
  const int t = threadIdx.x;

  v4f x0 = *(const v4f*)(s + (long)t * 4);
  v4f x1 = *(const v4f*)(s + (long)(t + 256) * 4);

  float m = x0[0];
  #pragma unroll
  for (int j = 1; j < 4; ++j) m = fmaxf(m, x0[j]);
  #pragma unroll
  for (int j = 0; j < 4; ++j) m = fmaxf(m, x1[j]);
  #pragma unroll
  for (int off = 16; off >= 1; off >>= 1) m = fmaxf(m, __shfl_xor(m, off, 32));

  __shared__ float redmax[8];
  __shared__ float redsum[8];
  if ((t & 31) == 0) redmax[t >> 5] = m;
  __syncthreads();
  float M = redmax[0];
  #pragma unroll
  for (int w = 1; w < 8; ++w) M = fmaxf(M, redmax[w]);

  float e0[4], e1[4], sum = 0.0f;
  #pragma unroll
  for (int j = 0; j < 4; ++j) { e0[j] = __expf(x0[j] - M); sum += e0[j]; }
  #pragma unroll
  for (int j = 0; j < 4; ++j) { e1[j] = __expf(x1[j] - M); sum += e1[j]; }
  #pragma unroll
  for (int off = 16; off >= 1; off >>= 1) sum += __shfl_xor(sum, off, 32);
  if ((t & 31) == 0) redsum[t >> 5] = sum;
  __syncthreads();
  float T = 0.0f;
  #pragma unroll
  for (int w = 0; w < 8; ++w) T += redsum[w];
  const float inv = 1.0f / T;

  v4h p0, p1;
  #pragma unroll
  for (int j = 0; j < 4; ++j) { p0[j] = (_Float16)(e0[j] * inv); p1[j] = (_Float16)(e1[j] * inv); }
  *(v4h*)(P + row * (long)n + (long)t * 4)         = p0;
  *(v4h*)(P + row * (long)n + (long)(t + 256) * 4) = p1;
}

extern "C" void kernel_launch(void* const* d_in, const int* in_sizes, int n_in,
                              void* d_out, int out_size, void* d_ws, size_t ws_size,
                              hipStream_t stream)
{
  const int B = 4, L = 2048, C = 512;
  const float* query = (const float*)d_in[0];
  const float* key_  = (const float*)d_in[1];
  const float* value = (const float*)d_in[2];
  const float* Wq = (const float*)d_in[3];
  const float* bq = (const float*)d_in[4];
  const float* Wk = (const float*)d_in[5];
  const float* bk = (const float*)d_in[6];
  const float* Wv = (const float*)d_in[7];
  const float* bv = (const float*)d_in[8];
  const float* Wo = (const float*)d_in[9];
  const float* bo = (const float*)d_in[10];

  // workspace partition (128 MB total)
  _Float16* qf  = (_Float16*)d_ws;                  // [B,L,C] f16
  _Float16* ktf = qf  + (size_t)B * L * C;          // [B,C,L] f16 (transposed)
  _Float16* vf  = ktf + (size_t)B * L * C;          // [B,L,C] f16
  _Float16* of  = vf  + (size_t)B * L * C;          // [B,L,C] f16
  float*    S   = (float*)(of + (size_t)B * L * C); // [B,L,L] f32
  _Float16* P   = (_Float16*)(S + (size_t)B * L * L);// [B,L,L] f16

  const float inv_sqrt_c = 1.0f / sqrtf((float)C);
  dim3 blk(256);
  dim3 gproj((B * L) / BM, C / BN, 1);  // (64, 4)

  // 1-3: input projections (fp32 x fp32 -> f16); K written pre-transposed
  gemm_wmma<0, 0, 1, 1><<<gproj, blk, 0, stream>>>(
      query, Wq, bq, qf,  C, C, C, C, 1.0f, 0, 0, 0, L);
  gemm_wmma<0, 0, 2, 1><<<gproj, blk, 0, stream>>>(
      key_,  Wk, bk, ktf, C, C, 0, C, 1.0f, 0, 0, (long)C * L, L);
  gemm_wmma<0, 0, 1, 1><<<gproj, blk, 0, stream>>>(
      value, Wv, bv, vf,  C, C, C, C, 1.0f, 0, 0, 0, L);

  // 4: scores S = scale * Q @ K^T  (batched; S stays L2-resident at 64 MB)
  dim3 gsc(L / BM, L / BN, B);  // (16, 16, 4)
  gemm_wmma<1, 1, 0, 0><<<gsc, blk, 0, stream>>>(
      qf, ktf, nullptr, S, C, L, L, C, inv_sqrt_c,
      (long)L * C, (long)C * L, (long)L * L, L);

  // 5: row softmax (fp32), emit P in f16
  softmax_rows<<<dim3(B * L), blk, 0, stream>>>(S, P, L);

  // 6: O = P @ V (batched, K = 2048)
  dim3 gpv(L / BM, C / BN, B);  // (16, 4, 4)
  gemm_wmma<1, 1, 1, 0><<<gpv, blk, 0, stream>>>(
      P, vf, nullptr, of, L, C, C, L, 1.0f,
      (long)L * L, (long)L * C, (long)L * C, L);

  // 7: output projection (f16 x fp32 -> fp32 d_out)
  gemm_wmma<1, 0, 0, 1><<<gproj, blk, 0, stream>>>(
      of, Wo, bo, (float*)d_out, C, C, C, C, 1.0f, 0, 0, 0, L);
}
